// SSA_34660386079274
// MI455X (gfx1250) — compile-verified, matrix operand order checked by reading, not run on hardware
//
#include <hip/hip_runtime.h>

// ---------------------------------------------------------------------------
// Types for CDNA5 WMMA (wave32): v16bf = A/B operand (16x32 / 32x16 bf16),
// v8f = 16x16 f32 C/D accumulator.
// ---------------------------------------------------------------------------
typedef __attribute__((ext_vector_type(16))) __bf16 v16bf;
typedef __attribute__((ext_vector_type(8)))  float  v8f;
typedef __attribute__((ext_vector_type(4)))  unsigned v4u;
typedef __attribute__((ext_vector_type(8)))  int      v8i;
typedef __attribute__((ext_vector_type(4)))  int      v4i;

// ISA 7.12.2: 16-bit A (16x32) / B (32x16) element->K mapping per lane.
// element i of the 16-element fragment maps to K = (i&7) + 16*(i>>3) + 8*(lane>=16)
__device__ __forceinline__ int kmap16(int i, int lane) {
    return (i & 7) + ((i >> 3) << 4) + ((lane >> 4) << 3);
}

__device__ __forceinline__ v8f wmma_bf16(const v16bf& a, const v16bf& b, const v8f& c) {
    return __builtin_amdgcn_wmma_f32_16x16x32_bf16(false, a, false, b, (short)0, c, false, false);
}

// ---------------------------------------------------------------------------
// Problem constants
// ---------------------------------------------------------------------------
#define BATCH 4
#define CCH   256
#define HW    4096   // 64*64
#define KVN   1024   // 32*32
#define NHEAD 8
#define DHEAD 32

// ---------------------------------------------------------------------------
// 1) 2x2 average pool: x[B,256,64,64] -> kv[B,256,32,32]
// ---------------------------------------------------------------------------
__global__ __launch_bounds__(256) void k_avgpool2(const float* __restrict__ x,
                                                  float* __restrict__ kv) {
    int i = blockIdx.x * 256 + threadIdx.x;     // over B*C*32*32 = 1,048,576
    int xo = i & 31;
    int yo = (i >> 5) & 31;
    int bc = i >> 10;
    const float* base = x + ((size_t)bc * 64 + yo * 2) * 64 + xo * 2;
    kv[i] = 0.25f * (base[0] + base[1] + base[64] + base[65]);
}

// ---------------------------------------------------------------------------
// 2) GroupNorm stats: one block per (b,g). Channels in a group are contiguous,
//    so each (b,g) reduces a contiguous span of `span` floats.
// ---------------------------------------------------------------------------
__global__ __launch_bounds__(256) void k_gn_stats(const float* __restrict__ X,
                                                  float* __restrict__ mu,
                                                  float* __restrict__ rs,
                                                  int span) {
    int bg = blockIdx.x;                         // 0..31 (b*8+g)
    const float* p = X + (size_t)bg * span;
    float s = 0.f, s2 = 0.f;
    for (int i = threadIdx.x; i < span; i += 256) {
        float v = p[i];
        s += v; s2 += v * v;
    }
    __shared__ float shs[256], shs2[256];
    shs[threadIdx.x] = s; shs2[threadIdx.x] = s2;
    __syncthreads();
    for (int off = 128; off > 0; off >>= 1) {
        if (threadIdx.x < off) {
            shs[threadIdx.x]  += shs[threadIdx.x + off];
            shs2[threadIdx.x] += shs2[threadIdx.x + off];
        }
        __syncthreads();
    }
    if (threadIdx.x == 0) {
        float m   = shs[0] / (float)span;
        float var = shs2[0] / (float)span - m * m;
        mu[bg] = m;
        rs[bg] = rsqrtf(var + 1e-5f);
    }
}

// ---------------------------------------------------------------------------
// 3) Fused GEMM: Out = W @ GroupNorm(X).  W:[256,256] f32, X:[B,256,Npix] f32.
//    The whole block shares one 16-pixel tile: the normalized B tile is staged
//    once per 32-channel K-chunk into LDS (transposed [pix][ch] so fragment
//    loads are K-contiguous -> ds_load_b128). 8 waves cover 128 out channels.
//    Output bf16 in [b, head, pix, 32] layout (ready for flash attention).
// ---------------------------------------------------------------------------
__global__ __launch_bounds__(256) void k_gemm_norm_qkv(
    const float* __restrict__ W, const float* __restrict__ X,
    const float* __restrict__ mu, const float* __restrict__ rs,
    const float* __restrict__ gamma, const float* __restrict__ beta,
    __bf16* __restrict__ Out, int Npix) {
    __shared__ __bf16 xt[16 * 32];               // [pix][ch_local]
    int lane = threadIdx.x & 31, wave = threadIdx.x >> 5;
    int b     = blockIdx.z;
    int mtile = blockIdx.y * 8 + wave;           // 16 M-tiles over 256 out ch
    int pix0  = blockIdx.x * 16;
    int col   = pix0 + (lane & 15);              // pixel (N) for D
    int rowA  = mtile * 16 + (lane & 15);        // out channel (M) for A-frag

    v8f acc = {};
    for (int kk = 0; kk < 8; ++kk) {
        int kb = kk * 32;
        __syncthreads();                         // xt reuse from previous chunk
        {   // cooperative, coalesced stage + normalize: 32 ch x 16 pix
            int pix = threadIdx.x & 15;
            int ch0 = threadIdx.x >> 4;          // 0..15
#pragma unroll
            for (int pass = 0; pass < 2; ++pass) {
                int ch = ch0 + pass * 16;        // local channel 0..31
                int k  = kb + ch;
                float xv  = X[((size_t)b * CCH + k) * Npix + pix0 + pix];
                int   g   = k >> 5;
                float sc  = rs[b * 8 + g] * gamma[k];
                float sh  = beta[k] - mu[b * 8 + g] * sc;
                xt[pix * 32 + ch] = (__bf16)(xv * sc + sh);
            }
        }
        __syncthreads();
        v16bf a, bf;
#pragma unroll
        for (int i = 0; i < 16; ++i) {
            int km = kmap16(i, lane);
            a[i]  = (__bf16)W[rowA * 256 + kb + km];   // contiguous -> b128
            bf[i] = xt[(lane & 15) * 32 + km];         // contiguous -> ds b128
        }
        acc = wmma_bf16(a, bf, acc);
    }
    // D layout: lane N=lane&15, VGPR r -> M = r + 8*(lane>=16)
    int rbase = (lane >> 4) << 3;
#pragma unroll
    for (int r = 0; r < 8; ++r) {
        int o = mtile * 16 + r + rbase;
        int head = o >> 5, dd = o & 31;
        Out[(((size_t)(b * NHEAD + head) * Npix) + col) * 32 + dd] = (__bf16)acc[r];
    }
}

// ---------------------------------------------------------------------------
// TDM: issue a 1-D tensor_load_to_lds of nelem contiguous bf16 elements from
// `gsrc` into LDS offset `ldsoff`. Descriptor per CDNA5 ISA ch.8.
// ---------------------------------------------------------------------------
#if defined(__has_builtin)
#if __has_builtin(__builtin_amdgcn_tensor_load_to_lds)
#define HAVE_TDM 1
#endif
#endif

#ifdef HAVE_TDM
__device__ __forceinline__ void tdm_load_1d(const void* gsrc, unsigned ldsoff,
                                            unsigned nelem) {
    unsigned long long ga = (unsigned long long)(uintptr_t)gsrc;
    // group0: count=1 | lds_addr | global_addr[56:0] | type=2 (bits 127:126)
    v4u g0 = { 1u, ldsoff, (unsigned)ga,
               (unsigned)((ga >> 32) & 0x01FFFFFFu) | 0x80000000u };
    // group1 bit layout (little-endian 32-bit words):
    unsigned w0 = 1u << 16;                     // data_size = 1 (2 bytes)
    unsigned w1 = (nelem & 0xFFFFu) << 16;      // tensor_dim0 [79:48] lo
    unsigned w2 = (nelem >> 16) | (1u << 16);   // tensor_dim0 hi | tensor_dim1=1
    unsigned w3 = (nelem & 0xFFFFu) << 16;      // tile_dim0 [127:112]
    unsigned w5 = nelem;                        // tensor_dim0_stride [207:160]
    v8i g1 = { (int)w0, (int)w1, (int)w2, (int)w3, 0, (int)w5, 0, 0 };
    v4i z4 = { 0, 0, 0, 0 };
#if __clang_major__ >= 23
    v8i z8 = { 0, 0, 0, 0, 0, 0, 0, 0 };
    __builtin_amdgcn_tensor_load_to_lds(g0, g1, z4, z4, z8, 0);
#else
    __builtin_amdgcn_tensor_load_to_lds(g0, g1, z4, z4, 0);
#endif
}
#endif

// ---------------------------------------------------------------------------
// 4) Flash attention. One wave owns 32 queries (two 16-row M-subtiles), so in
//    the online-softmax phase every lane of the wave32 owns exactly one query
//    row (no divergence). Per (b,h): Q[4096,32], K/V[1024,32] bf16.
//    64-key chunks: K staged by the Tensor Data Mover (contiguous 4KB tile),
//    V staged transposed ([d][key]) so O-WMMA B-frags are K-contiguous.
//    d == 32 == WMMA K, so each S tile is a single wmma. 1/sqrt(d) folded
//    into the Q fragment at load.
// ---------------------------------------------------------------------------
__global__ __launch_bounds__(256) void k_flash_attn(
    const __bf16* __restrict__ Qt, const __bf16* __restrict__ Kt,
    const __bf16* __restrict__ Vt, __bf16* __restrict__ Yt) {
    __shared__ __bf16 kbuf[64 * 32];      // [key][d]
    __shared__ __bf16 vbufT[32 * 64];     // [d][key]
    __shared__ float  sbuf[8][32 * 64];   // wave-private S tiles (f32)
    __shared__ __bf16 pbuf[8][32 * 64];   // wave-private P tiles (bf16)
    __shared__ float  fbuf[8][32];        // per-row rescale factors
    __shared__ float  lbuf[8][32];        // per-row 1/l

    int lane = threadIdx.x & 31, wave = threadIdx.x >> 5;
    int bh = blockIdx.y;                              // b*8+head
    int q0 = blockIdx.x * 256 + wave * 32;            // first query of wave
    const __bf16* Q = Qt + (size_t)bh * HW  * 32;
    const __bf16* K = Kt + (size_t)bh * KVN * 32;
    const __bf16* V = Vt + (size_t)bh * KVN * 32;

    const float scale = 0.17677669529663687f;         // 1/sqrt(32)

    // Q A-fragments: two [M=16, K=32=d] tiles, scale pre-folded
    v16bf aq[2];
#pragma unroll
    for (int mt = 0; mt < 2; ++mt) {
        int row = q0 + mt * 16 + (lane & 15);
#pragma unroll
        for (int i = 0; i < 16; ++i)
            aq[mt][i] = (__bf16)((float)Q[(size_t)row * 32 + kmap16(i, lane)] * scale);
    }

    v8f oacc[2][2] = {};                              // [mtile][dtile]
    float m_i = -1e30f, l_i = 0.f;
    int rbase = (lane >> 4) << 3;

    for (int kc = 0; kc < KVN / 64; ++kc) {
        __syncthreads();  // all waves done reading kbuf/vbufT of prev chunk
#ifdef HAVE_TDM
        if (wave == 0) {  // TDM: one 4KB contiguous K tile -> LDS
            tdm_load_1d(K + (size_t)kc * 64 * 32,
                        (unsigned)(uintptr_t)kbuf, 64 * 32);
            __builtin_amdgcn_s_wait_tensorcnt(0);
        }
#else
        {
            const unsigned* ksrc = (const unsigned*)(K + (size_t)kc * 64 * 32);
            unsigned* kdst = (unsigned*)kbuf;
            for (int t = threadIdx.x; t < 64 * 32 / 2; t += 256)
                kdst[t] = ksrc[t];
        }
#endif
        // stage V transposed: vbufT[d][key]
        for (int t = threadIdx.x; t < 64 * 32; t += 256) {
            int key = t >> 5, d = t & 31;
            vbufT[d * 64 + key] = V[(size_t)(kc * 64 + key) * 32 + d];
        }
        // prefetch next chunk into cache
        if (kc + 1 < KVN / 64) {
            __builtin_prefetch(K + (size_t)(kc + 1) * 64 * 32 + threadIdx.x * 8, 0, 0);
            __builtin_prefetch(V + (size_t)(kc + 1) * 64 * 32 + threadIdx.x * 8, 0, 0);
        }
        __syncthreads();

        // S = Q K^T : 4 key tiles x 2 M-subtiles, each one WMMA (K=32=d)
#pragma unroll
        for (int nt = 0; nt < 4; ++nt) {
            v16bf bk;
            int key = nt * 16 + (lane & 15);          // N = key column
#pragma unroll
            for (int i = 0; i < 16; ++i) bk[i] = kbuf[key * 32 + kmap16(i, lane)];
#pragma unroll
            for (int mt = 0; mt < 2; ++mt) {
                v8f s = {};
                s = wmma_bf16(aq[mt], bk, s);
#pragma unroll
                for (int r = 0; r < 8; ++r)
                    sbuf[wave][(mt * 16 + r + rbase) * 64 + nt * 16 + (lane & 15)] = s[r];
            }
        }
        // online softmax: every lane owns one of the wave's 32 query rows
        // (wave-private LDS, in-order per-wave LDS ops -> no block barrier)
        {
            float* srow = &sbuf[wave][lane * 64];
            float mnew = m_i;
#pragma unroll
            for (int j = 0; j < 64; ++j) mnew = fmaxf(mnew, srow[j]);
            float corr = __expf(m_i - mnew);
            float lsum = 0.f;
#pragma unroll
            for (int j = 0; j < 64; ++j) {
                float pv = __expf(srow[j] - mnew);
                pbuf[wave][lane * 64 + j] = (__bf16)pv;
                lsum += pv;
            }
            l_i = l_i * corr + lsum;
            m_i = mnew;
            fbuf[wave][lane] = corr;
        }
        // rescale running O accumulators (per-row factor via LDS broadcast)
#pragma unroll
        for (int mt = 0; mt < 2; ++mt)
#pragma unroll
            for (int r = 0; r < 8; ++r) {
                float c = fbuf[wave][mt * 16 + r + rbase];
                oacc[mt][0][r] *= c;
                oacc[mt][1][r] *= c;
            }
        // O += P V : two 32-key P chunks x two M-subtiles x two 16-wide d tiles
#pragma unroll
        for (int pc = 0; pc < 2; ++pc) {
            v16bf bv0, bv1;
            int dcol = lane & 15;
#pragma unroll
            for (int i = 0; i < 16; ++i) {
                int km = kmap16(i, lane);
                bv0[i] = vbufT[dcol * 64 + pc * 32 + km];        // contiguous
                bv1[i] = vbufT[(dcol + 16) * 64 + pc * 32 + km]; // contiguous
            }
#pragma unroll
            for (int mt = 0; mt < 2; ++mt) {
                v16bf ap;
                int prow = mt * 16 + (lane & 15);
#pragma unroll
                for (int i = 0; i < 16; ++i)
                    ap[i] = pbuf[wave][prow * 64 + pc * 32 + kmap16(i, lane)];
                oacc[mt][0] = wmma_bf16(ap, bv0, oacc[mt][0]);
                oacc[mt][1] = wmma_bf16(ap, bv1, oacc[mt][1]);
            }
        }
    }

    lbuf[wave][lane] = 1.f / l_i;
#pragma unroll
    for (int mt = 0; mt < 2; ++mt)
#pragma unroll
        for (int r = 0; r < 8; ++r) {
            float inv = lbuf[wave][mt * 16 + r + rbase];
            int qrow  = q0 + mt * 16 + r + rbase;
            int dcol  = lane & 15;
            Yt[((size_t)bh * HW + qrow) * 32 + dcol]      = (__bf16)(oacc[mt][0][r] * inv);
            Yt[((size_t)bh * HW + qrow) * 32 + 16 + dcol] = (__bf16)(oacc[mt][1][r] * inv);
        }
}

// ---------------------------------------------------------------------------
// 5) Projection + residual: out = x + Wproj @ y.  y is bf16 in [b,h,p,32];
//    K-chunks of 32 align exactly with heads, so B-fragment loads are
//    contiguous 8-element groups (b128) without any staging.
// ---------------------------------------------------------------------------
__global__ __launch_bounds__(256) void k_gemm_proj(
    const float* __restrict__ W, const __bf16* __restrict__ Yt,
    const float* __restrict__ Xres, float* __restrict__ Out) {
    int lane = threadIdx.x & 31, wave = threadIdx.x >> 5;
    int b     = blockIdx.z;
    int mtile = blockIdx.y * 8 + wave;
    int col   = blockIdx.x * 16 + (lane & 15);
    int rowA  = mtile * 16 + (lane & 15);

    v8f acc = {};
    for (int kk = 0; kk < 8; ++kk) {             // kk == head index
        const __bf16* ybase = Yt + (((size_t)(b * NHEAD + kk) * HW) + col) * 32;
        v16bf a, bf;
#pragma unroll
        for (int i = 0; i < 16; ++i) {
            int km = kmap16(i, lane);
            a[i]  = (__bf16)W[rowA * 256 + kk * 32 + km];
            bf[i] = ybase[km];
        }
        acc = wmma_bf16(a, bf, acc);
    }
    int rbase = (lane >> 4) << 3;
#pragma unroll
    for (int r = 0; r < 8; ++r) {
        int o = mtile * 16 + r + rbase;
        size_t idx = ((size_t)b * CCH + o) * HW + col;
        Out[idx] = Xres[idx] + acc[r];
    }
}

// ---------------------------------------------------------------------------
// Launch
// ---------------------------------------------------------------------------
extern "C" void kernel_launch(void* const* d_in, const int* in_sizes, int n_in,
                              void* d_out, int out_size, void* d_ws, size_t ws_size,
                              hipStream_t stream) {
    const float* x     = (const float*)d_in[0];
    const float* wq    = (const float*)d_in[1];
    const float* wk    = (const float*)d_in[2];
    const float* wv    = (const float*)d_in[3];
    const float* wproj = (const float*)d_in[4];
    const float* gnq_w = (const float*)d_in[5];
    const float* gnq_b = (const float*)d_in[6];
    const float* gnk_w = (const float*)d_in[7];
    const float* gnk_b = (const float*)d_in[8];
    float* out = (float*)d_out;

    char* ws = (char*)d_ws;
    float* kv  = (float*)ws;                                   // 4 MB
    float* muX = (float*)(ws + (4u << 20));
    float* rsX = muX + 32;
    float* muK = rsX + 32;
    float* rsK = muK + 32;
    __bf16* qt = (__bf16*)(ws + (4u << 20) + 4096);            // 8 MB
    __bf16* kt = qt + (size_t)BATCH * NHEAD * HW  * 32;        // 2 MB
    __bf16* vt = kt + (size_t)BATCH * NHEAD * KVN * 32;        // 2 MB
    __bf16* yt = vt + (size_t)BATCH * NHEAD * KVN * 32;        // 8 MB

    k_avgpool2<<<4096, 256, 0, stream>>>(x, kv);
    k_gn_stats<<<32, 256, 0, stream>>>(x,  muX, rsX, 32 * HW);
    k_gn_stats<<<32, 256, 0, stream>>>(kv, muK, rsK, 32 * KVN);

    k_gemm_norm_qkv<<<dim3(HW / 16, 2, BATCH), 256, 0, stream>>>(
        wq, x,  muX, rsX, gnq_w, gnq_b, qt, HW);
    k_gemm_norm_qkv<<<dim3(KVN / 16, 2, BATCH), 256, 0, stream>>>(
        wk, kv, muK, rsK, gnk_w, gnk_b, kt, KVN);
    k_gemm_norm_qkv<<<dim3(KVN / 16, 2, BATCH), 256, 0, stream>>>(
        wv, kv, muK, rsK, gnk_w, gnk_b, vt, KVN);

    k_flash_attn<<<dim3(HW / 256, BATCH * NHEAD), 256, 0, stream>>>(qt, kt, vt, yt);

    k_gemm_proj<<<dim3(HW / 16, 2, BATCH), 256, 0, stream>>>(wproj, yt, x, out);
}
